// CompressedGPT2Attention_43215960932849
// MI455X (gfx1250) — matrix operand order repaired and checked by reading, not compiled, not run on hardware
//
#include <hip/hip_runtime.h>

typedef __attribute__((ext_vector_type(16))) _Float16 v16h;
typedef __attribute__((ext_vector_type(8)))  _Float16 v8h;
typedef __attribute__((ext_vector_type(4)))  _Float16 v4h;
typedef __attribute__((ext_vector_type(2)))  _Float16 v2h;
typedef __attribute__((ext_vector_type(8)))  float    v8f;

__device__ __forceinline__ v8f wmma16(v16h a, v16h b, v8f c) {
    // D = A(16x32 f16) * B(32x16 f16) + C(16x16 f32)
    return __builtin_amdgcn_wmma_f32_16x16x32_f16(false, a, false, b, (short)0, c,
                                                  false, false);
}

// A-fragment (16x32): lane L<16 holds row L, K={0..7,16..23}; L>=16 row L-16, K={8..15,24..31}
__device__ __forceinline__ v16h load_frag_a(const _Float16* base, int ld) {
    const int lane = threadIdx.x & 31;
    const int half = lane >> 4;
    const _Float16* p = base + (lane & 15) * ld + half * 8;
    v16h v;
    ((v8h*)&v)[0] = *(const v8h*)(p);
    ((v8h*)&v)[1] = *(const v8h*)(p + 16);
    return v;
}

// B-fragment (32x16): lane L<16 holds col L, K={0..7,16..23}; L>=16 col L-16, K={8..15,24..31}.
// base points into a [col][k] (transposed) buffer so reads are contiguous.
__device__ __forceinline__ v16h load_frag_b(const _Float16* base, int ld) {
    const int lane = threadIdx.x & 31;
    const int half = lane >> 4;
    const _Float16* p = base + (lane & 15) * ld + half * 8;
    v16h v;
    ((v8h*)&v)[0] = *(const v8h*)(p);
    ((v8h*)&v)[1] = *(const v8h*)(p + 16);
    return v;
}

// Convert float4 pair -> v8h
__device__ __forceinline__ v8h cvt8(float4 a, float4 b) {
    v8h h;
    h[0] = (_Float16)a.x; h[1] = (_Float16)a.y; h[2] = (_Float16)a.z; h[3] = (_Float16)a.w;
    h[4] = (_Float16)b.x; h[5] = (_Float16)b.y; h[6] = (_Float16)b.z; h[7] = (_Float16)b.w;
    return h;
}

// C[M,N] = A[M,K] * B[K,N] + bias[N], f32 in/out, f16 WMMA compute.
// Block: 256 threads (8 waves). Block tile 128x256, wave tile 64x64, K-step 32.
// LDS double-buffered; global->reg fetch of tile k+1 overlaps WMMAs on tile k.
__global__ __launch_bounds__(256)
void gemm_bias_wmma(const float* __restrict__ A, const float* __restrict__ Bm,
                    const float* __restrict__ bias, float* __restrict__ C,
                    int M, int N, int K) {
    __shared__ _Float16 sA[2][128][32];   // 2 x 8 KB
    __shared__ _Float16 sB[2][256][32];   // 2 x 16 KB, transposed [n][k]
    const int tid  = threadIdx.x;
    const int lane = tid & 31;
    const int wave = tid >> 5;
    const int wm   = wave >> 2;        // 0..1 -> 64-row band
    const int wn   = wave & 3;         // 0..3 -> 64-col band
    const int m0   = blockIdx.y * 128;
    const int n0   = blockIdx.x * 256;

    v8f acc[4][4];
#pragma unroll
    for (int i = 0; i < 4; ++i)
#pragma unroll
        for (int j = 0; j < 4; ++j) acc[i][j] = {};

    const int arow = tid >> 1, ach = (tid & 1) * 16;        // A: 16 f32 per thread
    const int bc0  = (tid & 31) * 8, br0 = (tid >> 5) * 4;  // B: 4 rows x 8 cols

    // staged (already f16) register tile
    v8h a0, a1;
    v4h bw[8];

    auto fetch = [&](int k0) {
        const float4* asrc = (const float4*)(A + (size_t)(m0 + arow) * K + k0 + ach);
        float4 f0 = asrc[0], f1 = asrc[1], f2 = asrc[2], f3 = asrc[3];
        a0 = cvt8(f0, f1);
        a1 = cvt8(f2, f3);
        _Float16 r16[4][8];
#pragma unroll
        for (int r = 0; r < 4; ++r) {
            const float4* bsrc = (const float4*)(Bm + (size_t)(k0 + br0 + r) * N + n0 + bc0);
            float4 g0 = bsrc[0], g1 = bsrc[1];
            r16[r][0] = (_Float16)g0.x; r16[r][1] = (_Float16)g0.y;
            r16[r][2] = (_Float16)g0.z; r16[r][3] = (_Float16)g0.w;
            r16[r][4] = (_Float16)g1.x; r16[r][5] = (_Float16)g1.y;
            r16[r][6] = (_Float16)g1.z; r16[r][7] = (_Float16)g1.w;
        }
#pragma unroll
        for (int j = 0; j < 8; ++j) {
            v4h w;
            w[0] = r16[0][j]; w[1] = r16[1][j]; w[2] = r16[2][j]; w[3] = r16[3][j];
            bw[j] = w;
        }
    };
    auto commit = [&](int buf) {
        *(v8h*)&sA[buf][arow][ach]     = a0;
        *(v8h*)&sA[buf][arow][ach + 8] = a1;
#pragma unroll
        for (int j = 0; j < 8; ++j)
            *(v4h*)&sB[buf][bc0 + j][br0] = bw[j];
    };

    fetch(0);
    commit(0);
    int cur = 0;
    for (int k0 = 0; k0 < K; k0 += 32) {
        __syncthreads();                       // buf 'cur' ready; buf 'cur^1' free
        const bool more = (k0 + 32) < K;
        if (more) fetch(k0 + 32);              // overlaps with WMMAs below

        v16h af[4];
#pragma unroll
        for (int mt = 0; mt < 4; ++mt)
            af[mt] = load_frag_a(&sA[cur][wm * 64 + mt * 16][0], 32);
#pragma unroll
        for (int nt = 0; nt < 4; ++nt) {
            v16h bf = load_frag_b(&sB[cur][wn * 64 + nt * 16][0], 32);
#pragma unroll
            for (int mt = 0; mt < 4; ++mt)
                acc[mt][nt] = wmma16(af[mt], bf, acc[mt][nt]);
        }
        if (more) commit(cur ^ 1);
        cur ^= 1;
    }

    const int half = lane >> 4, ln = lane & 15;
#pragma unroll
    for (int nt = 0; nt < 4; ++nt) {
        const int col = n0 + wn * 64 + nt * 16 + ln;
        const float bv = bias[col];
#pragma unroll
        for (int mt = 0; mt < 4; ++mt) {
            const int r0 = m0 + wm * 64 + mt * 16 + half * 8;
#pragma unroll
            for (int i = 0; i < 8; ++i)
                C[(size_t)(r0 + i) * N + col] = acc[mt][nt][i] + bv;
        }
    }
}

// Flash attention + symmetric per-head projector.
// Grid: (S/64, H, B). Block: 128 threads (4 waves, 16 query rows each).
// K/V LDS double-buffered; next tile fetched to regs during WMMA/softmax.
__global__ __launch_bounds__(128)
void flash_attn_proj(const float* __restrict__ qkv, const float* __restrict__ projectors,
                     float* __restrict__ attnm) {
    __shared__ _Float16 sK[2][64][64];     // keys: [key][d]        2 x 8 KB
    __shared__ _Float16 sVt[2][64][64];    // values^T: [d][key]    2 x 8 KB
    __shared__ _Float16 sP[4][16][64];     // per-wave P / O relayout buffer
    __shared__ _Float16 sProj[64][64];     // symmetric projector as [e][d]
    const int S = 2048, D3 = 3072;
    const int qt = blockIdx.x, h = blockIdx.y, b = blockIdx.z;
    const int tid = threadIdx.x, lane = tid & 31, wave = tid >> 5;
    const int half = lane >> 4, ln = lane & 15;

    {   // stage projector (projectors are basis@basis^T -> symmetric): float4 + b64
        const float4* Pw = (const float4*)(projectors + (size_t)h * 4096);
        for (int i = tid; i < 1024; i += 128) {
            float4 f = Pw[i];
            v4h w;
            w[0] = (_Float16)f.x; w[1] = (_Float16)f.y;
            w[2] = (_Float16)f.z; w[3] = (_Float16)f.w;
            *(v4h*)&sProj[0][i * 4] = w;
        }
    }

    // Q A-fragments straight from global (rows are contiguous f32)
    v16h qf[2];
    {
        const float* qrow = qkv + (size_t)(b * S + qt * 64 + wave * 16 + ln) * D3 + h * 64;
#pragma unroll
        for (int kc = 0; kc < 2; ++kc) {
            const float4* p0 = (const float4*)(qrow + kc * 32 + half * 8);
            const float4* p1 = (const float4*)(qrow + kc * 32 + half * 8 + 16);
            v16h v;
            ((v8h*)&v)[0] = cvt8(p0[0], p0[1]);
            ((v8h*)&v)[1] = cvt8(p1[0], p1[1]);
            qf[kc] = v;
        }
    }

    float m[8], l[8];
    v8f o[4];
#pragma unroll
    for (int i = 0; i < 8; ++i) { m[i] = -3.0e38f; l[i] = 0.f; }
#pragma unroll
    for (int i = 0; i < 4; ++i) o[i] = {};

    const float* kbase = qkv + (size_t)(b * S) * D3 + 1024 + h * 64;
    const float* vbase = qkv + (size_t)(b * S) * D3 + 2048 + h * 64;
    const int kp = tid >> 2;          // 0..31 -> keys 2kp, 2kp+1
    const int dg = (tid & 3) * 16;    // d group of 16

    // staged (already f16) register K/V tile
    v8h kr[2][2];
    _Float16 vh[2][16];

    auto fetchKV = [&](int kt) {
#pragma unroll
        for (int kk = 0; kk < 2; ++kk) {
            const int keyi = kp * 2 + kk;
            const float4* ks = (const float4*)(kbase + (size_t)(kt * 64 + keyi) * D3 + dg);
            float4 k0 = ks[0], k1 = ks[1], k2 = ks[2], k3 = ks[3];
            kr[kk][0] = cvt8(k0, k1);
            kr[kk][1] = cvt8(k2, k3);
            const float4* vs = (const float4*)(vbase + (size_t)(kt * 64 + keyi) * D3 + dg);
            float4 v0 = vs[0], v1 = vs[1], v2 = vs[2], v3 = vs[3];
            vh[kk][0]  = (_Float16)v0.x; vh[kk][1]  = (_Float16)v0.y;
            vh[kk][2]  = (_Float16)v0.z; vh[kk][3]  = (_Float16)v0.w;
            vh[kk][4]  = (_Float16)v1.x; vh[kk][5]  = (_Float16)v1.y;
            vh[kk][6]  = (_Float16)v1.z; vh[kk][7]  = (_Float16)v1.w;
            vh[kk][8]  = (_Float16)v2.x; vh[kk][9]  = (_Float16)v2.y;
            vh[kk][10] = (_Float16)v2.z; vh[kk][11] = (_Float16)v2.w;
            vh[kk][12] = (_Float16)v3.x; vh[kk][13] = (_Float16)v3.y;
            vh[kk][14] = (_Float16)v3.z; vh[kk][15] = (_Float16)v3.w;
        }
        __builtin_prefetch(kbase + (size_t)((kt + 1) * 64 + kp * 2) * D3 + dg, 0, 0);
    };
    auto commitKV = [&](int buf) {
#pragma unroll
        for (int kk = 0; kk < 2; ++kk) {
            *(v8h*)&sK[buf][kp * 2 + kk][dg]     = kr[kk][0];
            *(v8h*)&sK[buf][kp * 2 + kk][dg + 8] = kr[kk][1];
        }
#pragma unroll
        for (int j = 0; j < 16; ++j) {
            v2h w;
            w[0] = vh[0][j]; w[1] = vh[1][j];
            *(v2h*)&sVt[buf][dg + j][kp * 2] = w;
        }
    };

    fetchKV(0);
    commitKV(0);
    int cur = 0;
    for (int kt = 0; kt <= qt; ++kt) {          // causal: tile-level skip
        __syncthreads();                        // buf 'cur' ready; 'cur^1' free
        const bool more = kt < qt;
        if (more) fetchKV(kt + 1);              // overlaps compute below

        // scores: 16x64 per wave = 4 col-blocks x 2 K-chunks of WMMA
        v8f s[4];
#pragma unroll
        for (int nb = 0; nb < 4; ++nb) {
            s[nb] = {};
            s[nb] = wmma16(qf[0], load_frag_b(&sK[cur][nb * 16][0], 64), s[nb]);
            s[nb] = wmma16(qf[1], load_frag_b(&sK[cur][nb * 16][32], 64), s[nb]);
        }
        const bool diag = (kt == qt);
#pragma unroll
        for (int nb = 0; nb < 4; ++nb) {
#pragma unroll
            for (int i = 0; i < 8; ++i) {
                float v = s[nb][i] * 0.125f;   // 1/sqrt(64)
                if (diag) {
                    const int row = qt * 64 + wave * 16 + i + half * 8;
                    const int col = kt * 64 + nb * 16 + ln;
                    if (col > row) v = -3.0e38f;
                }
                s[nb][i] = v;
            }
        }
        // online softmax (row = i + 8*half, spread over 16 lanes)
#pragma unroll
        for (int i = 0; i < 8; ++i) {
            float tm = fmaxf(fmaxf(s[0][i], s[1][i]), fmaxf(s[2][i], s[3][i]));
#pragma unroll
            for (int dd = 1; dd < 16; dd <<= 1) tm = fmaxf(tm, __shfl_xor(tm, dd, 32));
            const float mn = fmaxf(m[i], tm);
            const float alpha = __expf(m[i] - mn);
            m[i] = mn;
            float rs = 0.f;
#pragma unroll
            for (int nb = 0; nb < 4; ++nb) {
                const float p = __expf(s[nb][i] - mn);
                s[nb][i] = p;
                rs += p;
            }
#pragma unroll
            for (int dd = 1; dd < 16; dd <<= 1) rs += __shfl_xor(rs, dd, 32);
            l[i] = l[i] * alpha + rs;
#pragma unroll
            for (int db = 0; db < 4; ++db) o[db][i] = o[db][i] * alpha;
        }
        // C-layout P -> LDS -> A-layout fragments, then P@V
#pragma unroll
        for (int nb = 0; nb < 4; ++nb)
#pragma unroll
            for (int i = 0; i < 8; ++i)
                sP[wave][i + half * 8][nb * 16 + ln] = (_Float16)s[nb][i];
        v16h pf0 = load_frag_a(&sP[wave][0][0], 64);
        v16h pf1 = load_frag_a(&sP[wave][0][32], 64);
#pragma unroll
        for (int db = 0; db < 4; ++db) {
            o[db] = wmma16(pf0, load_frag_b(&sVt[cur][db * 16][0], 64), o[db]);
            o[db] = wmma16(pf1, load_frag_b(&sVt[cur][db * 16][32], 64), o[db]);
        }
        if (more) commitKV(cur ^ 1);
        cur ^= 1;
    }

    // normalize, apply projector via WMMA, store merged-head layout
    float inv[8];
#pragma unroll
    for (int i = 0; i < 8; ++i) inv[i] = 1.0f / l[i];
#pragma unroll
    for (int db = 0; db < 4; ++db)
#pragma unroll
        for (int i = 0; i < 8; ++i)
            sP[wave][i + half * 8][db * 16 + ln] = (_Float16)(o[db][i] * inv[i]);
    v16h of0 = load_frag_a(&sP[wave][0][0], 64);
    v16h of1 = load_frag_a(&sP[wave][0][32], 64);
#pragma unroll
    for (int eb = 0; eb < 4; ++eb) {
        v8f r = {};
        r = wmma16(of0, load_frag_b(&sProj[eb * 16][0], 64), r);
        r = wmma16(of1, load_frag_b(&sProj[eb * 16][32], 64), r);
        const int row0 = b * S + qt * 64 + wave * 16 + half * 8;
        const int col = h * 64 + eb * 16 + ln;
#pragma unroll
        for (int i = 0; i < 8; ++i)
            attnm[(size_t)(row0 + i) * 1024 + col] = r[i];
    }
}

extern "C" void kernel_launch(void* const* d_in, const int* in_sizes, int n_in,
                              void* d_out, int out_size, void* d_ws, size_t ws_size,
                              hipStream_t stream) {
    (void)in_sizes; (void)n_in; (void)out_size; (void)ws_size;
    const float* hidden   = (const float*)d_in[0];
    const float* c_attn_w = (const float*)d_in[1];
    const float* c_attn_b = (const float*)d_in[2];
    const float* c_proj_w = (const float*)d_in[3];
    const float* c_proj_b = (const float*)d_in[4];
    const float* proj     = (const float*)d_in[5];
    float* out  = (float*)d_out;
    float* qkv  = (float*)d_ws;                      // [4096,3072] f32 (48 MB)
    float* attn = qkv + (size_t)4096 * 3072;         // [4096,1024] f32 (16 MB)

    // 1) QKV = hidden @ c_attn_w + c_attn_b
    dim3 g1(3072 / 256, 4096 / 128);
    gemm_bias_wmma<<<g1, dim3(256), 0, stream>>>(hidden, c_attn_w, c_attn_b, qkv,
                                                 4096, 3072, 1024);
    // 2) flash attention + per-head projector -> merged heads
    dim3 g2(2048 / 64, 16, 2);
    flash_attn_proj<<<g2, dim3(128), 0, stream>>>(qkv, proj, attn);
    // 3) out = attn @ c_proj_w + c_proj_b
    dim3 g3(1024 / 256, 4096 / 128);
    gemm_bias_wmma<<<g3, dim3(256), 0, stream>>>(attn, c_proj_w, c_proj_b, out,
                                                 4096, 1024, 1024);
}
